// MSTRC_81758997447373
// MI455X (gfx1250) — compile-verified, matrix operand order checked by reading, not run on hardware
//
#include <hip/hip_runtime.h>
#include <hip/hip_bf16.h>
#include <math.h>

// ---------------------------------------------------------------------------
// Types for CDNA5 WMMA (wave32): v_wmma_f32_16x16x32_bf16
// ---------------------------------------------------------------------------
typedef __attribute__((ext_vector_type(16))) __bf16 v16bf;
typedef __attribute__((ext_vector_type(8)))  __bf16 v8bf;
typedef __attribute__((ext_vector_type(8)))  float  v8f;

#define N_NODES 20000
#define N_EDGES 320000
#define N_GRAPH 512

// ---------------------------------------------------------------------------
// float atomic max via integer punning (works for mixed signs; init = -inf)
// ---------------------------------------------------------------------------
__device__ __forceinline__ void atomic_max_f32(float* addr, float value) {
    if (value >= 0.0f) {
        atomicMax((int*)addr, __float_as_int(value));
    } else {
        atomicMin((unsigned int*)addr, __float_as_uint(value));
    }
}

// ---------------------------------------------------------------------------
// fill
// ---------------------------------------------------------------------------
__global__ void k_fill(float* p, float v, int n) {
    int i = blockIdx.x * blockDim.x + threadIdx.x;
    if (i < n) p[i] = v;
}

// ---------------------------------------------------------------------------
// fp32 activations [M x K] -> bf16 [M x Kp] (zero pad K..Kp)
// ---------------------------------------------------------------------------
__global__ void k_cvt_act(const float* __restrict__ src, __bf16* __restrict__ dst,
                          int M, int K, int Kp) {
    int i = blockIdx.x * blockDim.x + threadIdx.x;
    if (i >= M * Kp) return;
    int m = i / Kp, k = i - m * Kp;
    dst[i] = (k < K) ? (__bf16)src[(size_t)m * K + k] : (__bf16)0.0f;
}

// ---------------------------------------------------------------------------
// fp32 weight [K x Nout] -> bf16 transposed [Nout x Kp] (zero pad K..Kp)
// ---------------------------------------------------------------------------
__global__ void k_cvt_wT(const float* __restrict__ W, __bf16* __restrict__ dst,
                         int K, int Kp, int Nout) {
    int i = blockIdx.x * blockDim.x + threadIdx.x;
    if (i >= Nout * Kp) return;
    int n = i / Kp, k = i - n * Kp;
    dst[i] = (k < K) ? (__bf16)W[(size_t)k * Nout + n] : (__bf16)0.0f;
}

// ---------------------------------------------------------------------------
// Fragment load: 16 bf16 per lane = two contiguous 16B chunks
// (ISA 16-bit A 16x32 layout: lanes 0-15 K{0..7,16..23}, lanes 16-31
//  K{8..15,24..31}; the hh*8 base offset is applied by the caller)
// ---------------------------------------------------------------------------
__device__ __forceinline__ v16bf load_frag(const __bf16* p) {
    v8bf x0 = *(const v8bf*)p;
    v8bf x1 = *(const v8bf*)(p + 16);
    return __builtin_shufflevector(x0, x1, 0,1,2,3,4,5,6,7,8,9,10,11,12,13,14,15);
}

// ---------------------------------------------------------------------------
// WMMA GEMM:  C[M x Nout] = A[M x Kp](bf16) * BT[Nout x Kp](bf16)^T + bias
// One wave32 computes a 32x64 output tile: 2 A frags x 4 B frags ->
// 8 back-to-back v_wmma_f32_16x16x32_bf16 per 32-wide K chunk
// (1.5 b128 loads per WMMA instead of 4).
// Requires: M % 32 == 0, Nout % 64 == 0, Kp % 32 == 0.
// ---------------------------------------------------------------------------
__global__ void k_gemm_bf16(const __bf16* __restrict__ A,
                            const __bf16* __restrict__ BT,
                            const float* __restrict__ bias,
                            float* __restrict__ C,
                            int M, int Kp, int Nout, int relu)
{
    const int lane  = threadIdx.x & 31;
    const int wave  = blockIdx.x * (blockDim.x >> 5) + (threadIdx.x >> 5);
    const int tilesN = Nout >> 6;            // 64-wide tiles
    const int tiles  = (M >> 5) * tilesN;    // 32-tall tiles
    if (wave >= tiles) return;
    const int tm = wave / tilesN;
    const int tn = wave - tm * tilesN;

    const int hh = lane >> 4;      // half-wave selector
    const int l  = lane & 15;

    const __bf16* arow0 = A + (size_t)(tm * 32 + l) * Kp + hh * 8;
    const __bf16* arow1 = arow0 + (size_t)16 * Kp;
    const __bf16* brow0 = BT + (size_t)(tn * 64 + l) * Kp + hh * 8;
    const __bf16* brow1 = brow0 + (size_t)16 * Kp;
    const __bf16* brow2 = brow0 + (size_t)32 * Kp;
    const __bf16* brow3 = brow0 + (size_t)48 * Kp;

    v8f acc[2][4];
#pragma unroll
    for (int j = 0; j < 4; ++j) {
        const float bv = bias ? bias[tn * 64 + j * 16 + l] : 0.0f;
        const v8f z = { bv, bv, bv, bv, bv, bv, bv, bv };
        acc[0][j] = z;
        acc[1][j] = z;
    }

    for (int k0 = 0; k0 < Kp; k0 += 32) {
        v16bf a0 = load_frag(arow0 + k0);
        v16bf a1 = load_frag(arow1 + k0);
        v16bf b0 = load_frag(brow0 + k0);
        v16bf b1 = load_frag(brow1 + k0);
        v16bf b2 = load_frag(brow2 + k0);
        v16bf b3 = load_frag(brow3 + k0);
        acc[0][0] = __builtin_amdgcn_wmma_f32_16x16x32_bf16(false, a0, false, b0, (short)0, acc[0][0], false, false);
        acc[0][1] = __builtin_amdgcn_wmma_f32_16x16x32_bf16(false, a0, false, b1, (short)0, acc[0][1], false, false);
        acc[0][2] = __builtin_amdgcn_wmma_f32_16x16x32_bf16(false, a0, false, b2, (short)0, acc[0][2], false, false);
        acc[0][3] = __builtin_amdgcn_wmma_f32_16x16x32_bf16(false, a0, false, b3, (short)0, acc[0][3], false, false);
        acc[1][0] = __builtin_amdgcn_wmma_f32_16x16x32_bf16(false, a1, false, b0, (short)0, acc[1][0], false, false);
        acc[1][1] = __builtin_amdgcn_wmma_f32_16x16x32_bf16(false, a1, false, b1, (short)0, acc[1][1], false, false);
        acc[1][2] = __builtin_amdgcn_wmma_f32_16x16x32_bf16(false, a1, false, b2, (short)0, acc[1][2], false, false);
        acc[1][3] = __builtin_amdgcn_wmma_f32_16x16x32_bf16(false, a1, false, b3, (short)0, acc[1][3], false, false);
    }

    // D layout: VGPR r -> row (hh*8 + r), col l  (within each 16x16 sub-tile)
#pragma unroll
    for (int i = 0; i < 2; ++i) {
        float* crow = C + (size_t)(tm * 32 + i * 16 + hh * 8) * Nout + tn * 64 + l;
#pragma unroll
        for (int j = 0; j < 4; ++j) {
#pragma unroll
            for (int r = 0; r < 8; ++r) {
                float v = acc[i][j][r];
                if (relu) v = v > 0.0f ? v : 0.0f;
                crow[(size_t)r * Nout + j * 16] = v;
            }
        }
    }
}

// ---------------------------------------------------------------------------
// Edge attention logits: wave per edge; per head h: dot(q[dst],k[src]) over 128
// ---------------------------------------------------------------------------
__global__ void k_edge_alpha(const float* __restrict__ q, const float* __restrict__ kk,
                             const int* __restrict__ src, const int* __restrict__ dst,
                             float* __restrict__ alpha, float* __restrict__ amax, int E)
{
    const int lane = threadIdx.x & 31;
    const int e = blockIdx.x * (blockDim.x >> 5) + (threadIdx.x >> 5);
    if (e >= E) return;
    const int s = src[e], d = dst[e];
    const float4* qd = (const float4*)(q  + (size_t)d * 512);
    const float4* ks = (const float4*)(kk + (size_t)s * 512);
#pragma unroll
    for (int h = 0; h < 4; ++h) {
        float4 a = qd[h * 32 + lane];
        float4 b = ks[h * 32 + lane];
        float p = a.x * b.x + a.y * b.y + a.z * b.z + a.w * b.w;
#pragma unroll
        for (int off = 16; off > 0; off >>= 1)
            p += __shfl_down(p, off, 32);
        if (lane == 0) {
            float v = p * 0.08838834764831845f;   // 1/sqrt(128)
            alpha[(size_t)e * 4 + h] = v;
            atomic_max_f32(&amax[(size_t)d * 4 + h], v);
        }
    }
}

// ---------------------------------------------------------------------------
// ex = exp(alpha - amax[dst]); accumulate denominator
// ---------------------------------------------------------------------------
__global__ void k_edge_exp(float* __restrict__ alpha, const int* __restrict__ dst,
                           const float* __restrict__ amax, float* __restrict__ den, int E)
{
    int i = blockIdx.x * blockDim.x + threadIdx.x;
    if (i >= 4 * E) return;
    int e = i >> 2, h = i & 3;
    int d = dst[e];
    float am = amax[(size_t)d * 4 + h];
    if (!isfinite(am)) am = 0.0f;
    float ex = __expf(alpha[i] - am);
    alpha[i] = ex;
    atomicAdd(&den[(size_t)d * 4 + h], ex);
}

// ---------------------------------------------------------------------------
// agg[dst] += (ex/den) * v[src] : wave per (edge, head), 4 floats / lane
// ---------------------------------------------------------------------------
__global__ void k_edge_agg(const float* __restrict__ v, const float* __restrict__ alpha,
                           const float* __restrict__ den,
                           const int* __restrict__ src, const int* __restrict__ dst,
                           float* __restrict__ agg, int E)
{
    const int lane = threadIdx.x & 31;
    const int idx = blockIdx.x * (blockDim.x >> 5) + (threadIdx.x >> 5);
    if (idx >= 4 * E) return;
    const int e = idx >> 2, h = idx & 3;
    const int s = src[e], d = dst[e];
    const float dn = den[(size_t)d * 4 + h];
    const float w = alpha[idx] / fmaxf(dn, 1e-16f);
    const float4 vv = ((const float4*)(v + (size_t)s * 512))[h * 32 + lane];
    float* ap = agg + (size_t)d * 512 + h * 128 + lane * 4;
    atomicAdd(ap + 0, w * vv.x);
    atomicAdd(ap + 1, w * vv.y);
    atomicAdd(ap + 2, w * vv.z);
    atomicAdd(ap + 3, w * vv.w);
}

// ---------------------------------------------------------------------------
// Pooling: per (node, channel)
// ---------------------------------------------------------------------------
__global__ void k_pool_nodes(const float* __restrict__ h, const int* __restrict__ batch,
                             float* __restrict__ gmax, float* __restrict__ gsum,
                             float* __restrict__ cnt, int Nn)
{
    int i = blockIdx.x * blockDim.x + threadIdx.x;
    if (i >= Nn * 128) return;
    int n = i >> 7, c = i & 127;
    int b = batch[n];
    float v = h[i];
    atomic_max_f32(&gmax[(size_t)b * 128 + c], v);
    atomicAdd(&gsum[(size_t)b * 128 + c], v);
    if (c == 0) atomicAdd(&cnt[b], 1.0f);
}

__global__ void k_pool_final(const float* __restrict__ gmax, const float* __restrict__ gsum,
                             const float* __restrict__ cnt, float* __restrict__ g)
{
    int i = blockIdx.x * blockDim.x + threadIdx.x;
    if (i >= N_GRAPH * 128) return;
    int b = i >> 7, c = i & 127;
    float m = gmax[i];
    if (!isfinite(m)) m = 0.0f;
    g[(size_t)b * 256 + c]       = m;
    g[(size_t)b * 256 + 128 + c] = gsum[i] / fmaxf(cnt[b], 1.0f);
}

// ---------------------------------------------------------------------------
// Final 256 -> 1 projection: wave per graph row
// ---------------------------------------------------------------------------
__global__ void k_final(const float* __restrict__ t3, const float* __restrict__ Wf2,
                        const float* __restrict__ bf2, float* __restrict__ out)
{
    const int lane = threadIdx.x & 31;
    const int b = blockIdx.x * (blockDim.x >> 5) + (threadIdx.x >> 5);
    if (b >= N_GRAPH) return;
    const float* row = t3 + (size_t)b * 256;
    float p = 0.0f;
#pragma unroll
    for (int j = 0; j < 8; ++j) p += row[lane + j * 32] * Wf2[lane + j * 32];
#pragma unroll
    for (int off = 16; off > 0; off >>= 1) p += __shfl_down(p, off, 32);
    if (lane == 0) out[b] = p + bf2[0];
}

// ---------------------------------------------------------------------------
// Launcher
// ---------------------------------------------------------------------------
extern "C" void kernel_launch(void* const* d_in, const int* in_sizes, int n_in,
                              void* d_out, int out_size, void* d_ws, size_t ws_size,
                              hipStream_t stream)
{
    (void)in_sizes; (void)n_in; (void)out_size; (void)ws_size;
    const int N = N_NODES, E = N_EDGES, B = N_GRAPH;

    const float* x    = (const float*)d_in[0];
    const int* ei     = (const int*)d_in[1];
    const int* src    = ei;        // edge_index[0]
    const int* dstp   = ei + E;    // edge_index[1]
    const int* batch  = (const int*)d_in[2];

    struct Layer { const float *Wq,*bq,*Wk,*bk,*Wv,*bv,*Ws,*bs,*Wm,*bm; };
    Layer L[3];
    int ii = 3;
    for (int l = 0; l < 3; ++l) {
        L[l].Wq = (const float*)d_in[ii++]; L[l].bq = (const float*)d_in[ii++];
        L[l].Wk = (const float*)d_in[ii++]; L[l].bk = (const float*)d_in[ii++];
        L[l].Wv = (const float*)d_in[ii++]; L[l].bv = (const float*)d_in[ii++];
        L[l].Ws = (const float*)d_in[ii++]; L[l].bs = (const float*)d_in[ii++];
        L[l].Wm = (const float*)d_in[ii++]; L[l].bm = (const float*)d_in[ii++];
    }
    const float* Wg1 = (const float*)d_in[ii++]; const float* bg1 = (const float*)d_in[ii++];
    const float* Wg2 = (const float*)d_in[ii++]; const float* bg2 = (const float*)d_in[ii++];
    const float* Wf1 = (const float*)d_in[ii++]; const float* bf1 = (const float*)d_in[ii++];
    const float* Wf2 = (const float*)d_in[ii++]; const float* bf2 = (const float*)d_in[ii++];

    // ---- workspace bump allocator -------------------------------------
    char* wp = (char*)d_ws;
    auto alloc = [&](size_t bytes) -> void* {
        void* r = (void*)wp;
        wp += (bytes + 255) & ~(size_t)255;
        return r;
    };
    float*  bufA  = (float*)alloc((size_t)N * 512 * 4);   // q, then v
    float*  bufB  = (float*)alloc((size_t)N * 512 * 4);   // k, then skip/agg
    float*  hbuf  = (float*)alloc((size_t)N * 128 * 4);   // layer output
    float*  alpha = (float*)alloc((size_t)E * 4 * 4);
    float*  amax  = (float*)alloc((size_t)N * 4 * 4);
    float*  den   = (float*)alloc((size_t)N * 4 * 4);
    __bf16* xb    = (__bf16*)alloc((size_t)N * 512 * 2);  // bf16 activation staging
    __bf16* wb    = (__bf16*)alloc((size_t)1024 * 1024 * 2); // bf16 weight^T staging
    float*  gmax  = (float*)alloc((size_t)B * 128 * 4);
    float*  gsum  = (float*)alloc((size_t)B * 128 * 4);
    float*  cnt   = (float*)alloc((size_t)B * 4);
    float*  g     = (float*)alloc((size_t)B * 256 * 4);
    float*  t1    = (float*)alloc((size_t)B * 1024 * 4);
    float*  t2    = (float*)alloc((size_t)B * 512 * 4);
    float*  t3    = (float*)alloc((size_t)B * 256 * 4);
    __bf16* gb    = (__bf16*)alloc((size_t)B * 1024 * 2); // MLP activation staging

    auto cvt_act = [&](const float* s, __bf16* d, int M, int K, int Kp) {
        int n = M * Kp;
        k_cvt_act<<<(n + 255) / 256, 256, 0, stream>>>(s, d, M, K, Kp);
    };
    auto cvt_wT = [&](const float* W, __bf16* d, int K, int Kp, int Nout) {
        int n = Nout * Kp;
        k_cvt_wT<<<(n + 255) / 256, 256, 0, stream>>>(W, d, K, Kp, Nout);
    };
    auto gemm = [&](const __bf16* A, const __bf16* BT, const float* bias, float* C,
                    int M, int Kp, int Nout, int relu) {
        int tiles = (M / 32) * (Nout / 64);
        k_gemm_bf16<<<(tiles + 7) / 8, 256, 0, stream>>>(A, BT, bias, C, M, Kp, Nout, relu);
    };
    auto fill = [&](float* p, float v, int n) {
        k_fill<<<(n + 255) / 256, 256, 0, stream>>>(p, v, n);
    };

    // ---- 3 TransformerConv layers -------------------------------------
    for (int l = 0; l < 3; ++l) {
        const float* hin = (l == 0) ? x : hbuf;
        const int K  = (l == 0) ? 43 : 128;
        const int Kp = (l == 0) ? 64 : 128;

        cvt_act(hin, xb, N, K, Kp);

        // q, k projections
        cvt_wT(L[l].Wq, wb, K, Kp, 512);  gemm(xb, wb, L[l].bq, bufA, N, Kp, 512, 0);
        cvt_wT(L[l].Wk, wb, K, Kp, 512);  gemm(xb, wb, L[l].bk, bufB, N, Kp, 512, 0);

        // segment softmax over incoming edges
        fill(amax, -__builtin_inff(), N * 4);
        fill(den, 0.0f, N * 4);
        k_edge_alpha<<<(E + 7) / 8, 256, 0, stream>>>(bufA, bufB, src, dstp, alpha, amax, E);
        k_edge_exp<<<(4 * E + 255) / 256, 256, 0, stream>>>(alpha, dstp, amax, den, E);

        // v projection (reuse bufA), skip connection into agg (bufB)
        cvt_wT(L[l].Wv, wb, K, Kp, 512);  gemm(xb, wb, L[l].bv, bufA, N, Kp, 512, 0);
        cvt_wT(L[l].Ws, wb, K, Kp, 512);  gemm(xb, wb, L[l].bs, bufB, N, Kp, 512, 0);

        // weighted aggregation: agg[dst] += w * v[src]
        k_edge_agg<<<(4 * E + 7) / 8, 256, 0, stream>>>(bufA, alpha, den, src, dstp, bufB, E);

        // h = relu(agg @ Wm + bm)
        cvt_act(bufB, xb, N, 512, 512);
        cvt_wT(L[l].Wm, wb, 512, 512, 128);
        gemm(xb, wb, L[l].bm, hbuf, N, 512, 128, 1);
    }

    // ---- global max + mean pooling ------------------------------------
    fill(gmax, -__builtin_inff(), B * 128);
    fill(gsum, 0.0f, B * 128);
    fill(cnt, 0.0f, B);
    k_pool_nodes<<<(N * 128 + 255) / 256, 256, 0, stream>>>(hbuf, batch, gmax, gsum, cnt, N);
    k_pool_final<<<(B * 128 + 255) / 256, 256, 0, stream>>>(gmax, gsum, cnt, g);

    // ---- MLP head ------------------------------------------------------
    cvt_act(g,  gb, B, 256, 256);   cvt_wT(Wg1, wb, 256, 256, 1024);
    gemm(gb, wb, bg1, t1, B, 256, 1024, 1);
    cvt_act(t1, gb, B, 1024, 1024); cvt_wT(Wg2, wb, 1024, 1024, 512);
    gemm(gb, wb, bg2, t2, B, 1024, 512, 1);
    cvt_act(t2, gb, B, 512, 512);   cvt_wT(Wf1, wb, 512, 512, 256);
    gemm(gb, wb, bf1, t3, B, 512, 256, 1);

    k_final<<<(B + 7) / 8, 256, 0, stream>>>(t3, Wf2, bf2, (float*)d_out);
}
